// MyModel_2911987826806
// MI455X (gfx1250) — compile-verified, matrix operand order checked by reading, not run on hardware
//
#include <hip/hip_runtime.h>

// ---------------------------------------------------------------------------
// Types for CDNA5 WMMA
// ---------------------------------------------------------------------------
typedef __attribute__((ext_vector_type(16))) __bf16 v16bf;
typedef __attribute__((ext_vector_type(8)))  float  v8f;

union FragBF { v16bf v; unsigned u[8]; };

__device__ __forceinline__ unsigned short f2bf(float f) {
    unsigned x = __float_as_uint(f);
    unsigned r = x + 0x7FFFu + ((x >> 16) & 1u);   // round-to-nearest-even
    return (unsigned short)(r >> 16);
}
__device__ __forceinline__ float bf2f(unsigned short u) {
    return __uint_as_float(((unsigned)u) << 16);
}

// ---------------------------------------------------------------------------
// Generic GEMM: C[M,N] = A[M,K] * B[N,K]^T (+ bias[N]) (optional relu)
// A: fp32 or bf16 (ABF16), B: fp32 weights, C: fp32 or bf16 (OUTBF16).
// Block tile 128x128, BK=32, 256 threads = 8 wave32 waves.
// Wave grid 4(M) x 2(N); each wave owns a 32x64 sub-tile = 8 accumulators,
// issuing 8 v_wmma_f32_16x16x32_bf16 per k-step (2 A-frags x 4 B-frags).
// Uniform full-tile fast path (no per-lane guards); prefetch of next k-tile.
// ---------------------------------------------------------------------------
#define BM 128
#define BN 128
#define BK 32
#define SA 40            // padded LDS row stride (bf16 elements)
#define GEMM_THREADS 256

template <bool ABF16, bool HASBIAS, bool RELU, bool OUTBF16, bool GUARD>
__device__ __forceinline__
void gemm_body(const void* __restrict__ Av, const float* __restrict__ B,
               const float* __restrict__ bias, void* __restrict__ Cv,
               int M, int N, int K, int lda, int ldb, int ldc,
               unsigned short* As, unsigned short* Bs)
{
    const int m0   = blockIdx.y * BM;
    const int n0   = blockIdx.x * BN;
    const int tid  = threadIdx.x;
    const int lane = tid & 31;
    const int wave = tid >> 5;
    const int wm   = wave & 3;          // 32-row strip within tile
    const int wn   = wave >> 2;         // 64-col half
    const int half = lane >> 4;
    const int r16  = lane & 15;

    v8f zero8 = {0.f,0.f,0.f,0.f,0.f,0.f,0.f,0.f};
    v8f acc[2][4];
    #pragma unroll
    for (int i = 0; i < 2; ++i)
        #pragma unroll
        for (int j = 0; j < 4; ++j) acc[i][j] = zero8;

    const int rowL = tid >> 2;          // 0..63   (loader row base)
    const int colL = (tid & 3) * 8;     // 0,8,16,24

    for (int k0 = 0; k0 < K; k0 += BK) {
        // ---- stage A tile (128 x 32) into LDS as bf16 (2 rows/thread) ----
        #pragma unroll
        for (int r = 0; r < 2; ++r) {
            int rowA = rowL + r * 64;
            int gm = m0 + rowA;
            unsigned short* dst = &As[rowA * SA + colL];
            if (!GUARD || gm < M) {
                if (ABF16) {
                    const unsigned short* ap =
                        (const unsigned short*)Av + (size_t)gm * lda + (k0 + colL);
                    *(uint4*)dst = *(const uint4*)ap;
                    if (k0 + BK < K) __builtin_prefetch(ap + BK, 0, 1);
                } else {
                    const float* ap = (const float*)Av + (size_t)gm * lda + (k0 + colL);
                    float4 t0 = *(const float4*)(ap);
                    float4 t1 = *(const float4*)(ap + 4);
                    dst[0] = f2bf(t0.x); dst[1] = f2bf(t0.y);
                    dst[2] = f2bf(t0.z); dst[3] = f2bf(t0.w);
                    dst[4] = f2bf(t1.x); dst[5] = f2bf(t1.y);
                    dst[6] = f2bf(t1.z); dst[7] = f2bf(t1.w);
                    if (k0 + BK < K) __builtin_prefetch(ap + BK, 0, 1);
                }
            } else {
                uint4 z = {0u, 0u, 0u, 0u};
                *(uint4*)dst = z;
            }
        }
        // ---- stage B tile (128 x 32) into LDS as bf16 (2 rows/thread) ----
        #pragma unroll
        for (int r = 0; r < 2; ++r) {
            int rowB = rowL + r * 64;
            int gn = n0 + rowB;
            unsigned short* dst = &Bs[rowB * SA + colL];
            if (!GUARD || gn < N) {
                const float* bp = B + (size_t)gn * ldb + (k0 + colL);
                float4 t0 = *(const float4*)(bp);
                float4 t1 = *(const float4*)(bp + 4);
                dst[0] = f2bf(t0.x); dst[1] = f2bf(t0.y);
                dst[2] = f2bf(t0.z); dst[3] = f2bf(t0.w);
                dst[4] = f2bf(t1.x); dst[5] = f2bf(t1.y);
                dst[6] = f2bf(t1.z); dst[7] = f2bf(t1.w);
                if (k0 + BK < K) __builtin_prefetch(bp + BK, 0, 1);
            } else {
                uint4 z = {0u, 0u, 0u, 0u};
                *(uint4*)dst = z;
            }
        }
        __syncthreads();

        // ---- A fragments: lane = row (M), element e -> K per ISA 7.12.2 ----
        FragBF fa[2];
        #pragma unroll
        for (int i = 0; i < 2; ++i) {
            const unsigned short* arow = &As[(wm * 32 + i * 16 + r16) * SA];
            #pragma unroll
            for (int d = 0; d < 8; ++d) {
                int kb = ((d & 4) << 2) + ((d & 3) << 1) + half * 8;  // {0,2,4,6,16,18,20,22}+8*half
                fa[i].u[d] = *(const unsigned*)(arow + kb);
            }
        }
        // ---- 8 WMMAs: B fragment per 16-col subtile (lane = col N) ----
        #pragma unroll
        for (int j = 0; j < 4; ++j) {
            FragBF fb;
            const unsigned short* brow = &Bs[(wn * 64 + j * 16 + r16) * SA];
            #pragma unroll
            for (int d = 0; d < 8; ++d) {
                int kb = ((d & 4) << 2) + ((d & 3) << 1) + half * 8;
                fb.u[d] = *(const unsigned*)(brow + kb);
            }
            acc[0][j] = __builtin_amdgcn_wmma_f32_16x16x32_bf16(
                            false, fa[0].v, false, fb.v, (short)0, acc[0][j], false, false);
            acc[1][j] = __builtin_amdgcn_wmma_f32_16x16x32_bf16(
                            false, fa[1].v, false, fb.v, (short)0, acc[1][j], false, false);
        }
        __syncthreads();
    }

    // ---- epilogue: C/D layout (lane = N, vgpr v -> M = v + 8*half) ----
    #pragma unroll
    for (int i = 0; i < 2; ++i) {
        #pragma unroll
        for (int j = 0; j < 4; ++j) {
            int gn = n0 + wn * 64 + j * 16 + r16;
            if (GUARD && gn >= N) continue;
            float bv = HASBIAS ? bias[gn] : 0.0f;
            #pragma unroll
            for (int v = 0; v < 8; ++v) {
                int gm = m0 + wm * 32 + i * 16 + half * 8 + v;
                if (GUARD && gm >= M) continue;
                float val = acc[i][j][v] + bv;
                if (RELU) val = fmaxf(val, 0.0f);
                if (OUTBF16)
                    ((unsigned short*)Cv)[(size_t)gm * ldc + gn] = f2bf(val);
                else
                    ((float*)Cv)[(size_t)gm * ldc + gn] = val;
            }
        }
    }
}

template <bool ABF16, bool HASBIAS, bool RELU, bool OUTBF16>
__global__ __launch_bounds__(GEMM_THREADS)
void gemm_wmma(const void* __restrict__ Av, const float* __restrict__ B,
               const float* __restrict__ bias, void* __restrict__ Cv,
               int M, int N, int K, int lda, int ldb, int ldc)
{
    __shared__ unsigned short sm[(BM + BN) * SA];
    unsigned short* As = sm;
    unsigned short* Bs = sm + BM * SA;
    // block-uniform branch: interior tiles take the unguarded fast path
    if ((blockIdx.y * BM + BM <= M) && (blockIdx.x * BN + BN <= N))
        gemm_body<ABF16, HASBIAS, RELU, OUTBF16, false>(Av, B, bias, Cv,
                                                        M, N, K, lda, ldb, ldc, As, Bs);
    else
        gemm_body<ABF16, HASBIAS, RELU, OUTBF16, true>(Av, B, bias, Cv,
                                                       M, N, K, lda, ldb, ldc, As, Bs);
}

template <bool ABF16, bool HASBIAS, bool RELU, bool OUTBF16>
static void launch_gemm(const void* A, const float* B, const float* bias, void* C,
                        int M, int N, int K, int lda, int ldb, int ldc, hipStream_t s)
{
    dim3 g((N + BN - 1) / BN, (M + BM - 1) / BM);
    gemm_wmma<ABF16, HASBIAS, RELU, OUTBF16><<<g, GEMM_THREADS, 0, s>>>(
        A, B, bias, C, M, N, K, lda, ldb, ldc);
}

// ---------------------------------------------------------------------------
// GAT helper kernels (8 heads x 32 dims, D_H = 256)
// ---------------------------------------------------------------------------
__device__ __forceinline__ float leaky(float x) { return x > 0.f ? x : 0.2f * x; }

__device__ __forceinline__ void atomicMaxF(float* addr, float v) {
    if (v >= 0.f) atomicMax((int*)addr, __float_as_int(v));
    else          atomicMin((unsigned int*)addr, __float_as_uint(v));
}

__global__ void gat_eler(const float* __restrict__ H, const float* __restrict__ al,
                         const float* __restrict__ ar, float* __restrict__ el,
                         float* __restrict__ er, int Nn)
{
    int i = blockIdx.x * blockDim.x + threadIdx.x;
    if (i >= Nn * 8) return;
    int n = i >> 3, h = i & 7;
    const float* hr = H + (size_t)n * 256 + h * 32;
    float sl = 0.f, sr = 0.f;
    #pragma unroll
    for (int d = 0; d < 32; ++d) {
        float x = hr[d];
        sl += x * al[h * 32 + d];
        sr += x * ar[h * 32 + d];
    }
    el[i] = sl; er[i] = sr;
}

__global__ void gat_init_max(const float* __restrict__ el, const float* __restrict__ er,
                             float* __restrict__ m, int Nn)
{
    int i = blockIdx.x * blockDim.x + threadIdx.x;
    if (i >= Nn * 8) return;
    m[i] = leaky(el[i] + er[i]);          // self-loop seeds the max
}

__global__ void gat_edge_max(const int* __restrict__ src, const int* __restrict__ dst,
                             int dstoff, const float* __restrict__ el,
                             const float* __restrict__ er, float* __restrict__ m, int E)
{
    int i = blockIdx.x * blockDim.x + threadIdx.x;
    if (i >= E * 8) return;
    int e = i >> 3, h = i & 7;
    int s = src[e], d = dst[e] + dstoff;
    float x = leaky(el[s * 8 + h] + er[d * 8 + h]);
    atomicMaxF(&m[d * 8 + h], x);
}

__global__ void gat_init_sum(const float* __restrict__ el, const float* __restrict__ er,
                             const float* __restrict__ m, const float* __restrict__ H,
                             float* __restrict__ z, float* __restrict__ agg, int Nn)
{
    int i = blockIdx.x * blockDim.x + threadIdx.x;
    if (i >= Nn * 8) return;
    int n = i >> 3, h = i & 7;
    float a = __expf(leaky(el[i] + er[i]) - m[i]);   // self-loop term
    z[i] = a;
    const float* hr = H + (size_t)n * 256 + h * 32;
    float* ag = agg + (size_t)n * 256 + h * 32;
    #pragma unroll
    for (int d = 0; d < 32; ++d) ag[d] = a * hr[d];
}

__global__ void gat_edge_acc(const int* __restrict__ src, const int* __restrict__ dst,
                             int dstoff, const float* __restrict__ el,
                             const float* __restrict__ er, const float* __restrict__ m,
                             const float* __restrict__ H, float* __restrict__ z,
                             float* __restrict__ agg, int E)
{
    int i = blockIdx.x * blockDim.x + threadIdx.x;
    if (i >= E * 8) return;
    int e = i >> 3, h = i & 7;
    int s = src[e], d = dst[e] + dstoff;
    float a = __expf(leaky(el[s * 8 + h] + er[d * 8 + h]) - m[d * 8 + h]);
    atomicAdd(&z[d * 8 + h], a);
    const float* hs = H + (size_t)s * 256 + h * 32;
    float* ag = agg + (size_t)d * 256 + h * 32;
    #pragma unroll
    for (int j = 0; j < 32; ++j) atomicAdd(&ag[j], a * hs[j]);
}

__global__ void gat_final(const float* __restrict__ agg, const float* __restrict__ z,
                          const float* __restrict__ bgat, float* __restrict__ tgt,
                          int rowoff, int slot, int NPn)
{
    int i = blockIdx.x * blockDim.x + threadIdx.x;
    if (i >= NPn * 8) return;
    int n = i >> 3, h = i & 7;
    int row = n + rowoff;
    float zi = 1.0f / z[row * 8 + h];
    const float* ag = agg + (size_t)row * 256 + h * 32;
    float* o = tgt + ((size_t)n * 3 + slot) * 256 + h * 32;
    #pragma unroll
    for (int j = 0; j < 32; ++j) o[j] = ag[j] * zi + bgat[h * 32 + j];
}

__global__ void copy_slot0(const float* __restrict__ hp, float* __restrict__ tgt, int NPn)
{
    int i = blockIdx.x * blockDim.x + threadIdx.x;
    if (i >= NPn * 256) return;
    int n = i >> 8, d = i & 255;
    tgt[((size_t)n * 3) * 256 + d] = hp[i];
}

// ---------------------------------------------------------------------------
// Tiny S=3 attention: one thread per (node, head, query-pos). qkv in bf16.
// ---------------------------------------------------------------------------
__global__ void attn_s3(const unsigned short* __restrict__ qp, int ldq,
                        const unsigned short* __restrict__ kp, int ldk,
                        const unsigned short* __restrict__ vp, int ldv,
                        float* __restrict__ out, int nodes)
{
    int i = blockIdx.x * blockDim.x + threadIdx.x;
    if (i >= nodes * 24) return;
    int s = i % 3;
    int t2 = i / 3;
    int h = t2 & 7;
    int n = t2 >> 3;
    const float scale = 0.1767766953f;   // 1/sqrt(32)

    float q[32];
    const unsigned short* qr = qp + (size_t)(n * 3 + s) * ldq + h * 32;
    #pragma unroll
    for (int d = 0; d < 32; ++d) q[d] = bf2f(qr[d]);

    float lg[3];
    #pragma unroll
    for (int t = 0; t < 3; ++t) {
        const unsigned short* kr = kp + (size_t)(n * 3 + t) * ldk + h * 32;
        float acc = 0.f;
        #pragma unroll
        for (int d = 0; d < 32; ++d) acc += q[d] * bf2f(kr[d]);
        lg[t] = acc * scale;
    }
    float mx = fmaxf(lg[0], fmaxf(lg[1], lg[2]));
    float e0 = __expf(lg[0] - mx), e1 = __expf(lg[1] - mx), e2 = __expf(lg[2] - mx);
    float inv = 1.0f / (e0 + e1 + e2);
    float a0 = e0 * inv, a1 = e1 * inv, a2 = e2 * inv;

    const unsigned short* v0 = vp + (size_t)(n * 3 + 0) * ldv + h * 32;
    const unsigned short* v1 = vp + (size_t)(n * 3 + 1) * ldv + h * 32;
    const unsigned short* v2 = vp + (size_t)(n * 3 + 2) * ldv + h * 32;
    float* orow = out + (size_t)(n * 3 + s) * 256 + h * 32;
    #pragma unroll
    for (int d = 0; d < 32; ++d)
        orow[d] = a0 * bf2f(v0[d]) + a1 * bf2f(v1[d]) + a2 * bf2f(v2[d]);
}

// ---------------------------------------------------------------------------
// Residual add + LayerNorm over 256 features: one wave32 per row.
// ---------------------------------------------------------------------------
__global__ __launch_bounds__(256)
void ln_add(const float* __restrict__ x, const float* __restrict__ r,
            const float* __restrict__ g, const float* __restrict__ b,
            float* __restrict__ out, int rows)
{
    int row = blockIdx.x * 8 + (threadIdx.x >> 5);
    if (row >= rows) return;
    int lane = threadIdx.x & 31;
    const float* xr = x + (size_t)row * 256;
    const float* rr = r + (size_t)row * 256;
    float v[8];
    float s = 0.f, sq = 0.f;
    #pragma unroll
    for (int j = 0; j < 8; ++j) {
        float t = xr[lane * 8 + j] + rr[lane * 8 + j];
        v[j] = t; s += t; sq += t * t;
    }
    #pragma unroll
    for (int o = 16; o > 0; o >>= 1) {
        s  += __shfl_xor(s, o, 32);
        sq += __shfl_xor(sq, o, 32);
    }
    float mu   = s * (1.0f / 256.0f);
    float var  = sq * (1.0f / 256.0f) - mu * mu;
    float rstd = rsqrtf(var + 1e-5f);
    float* orow = out + (size_t)row * 256;
    #pragma unroll
    for (int j = 0; j < 8; ++j) {
        int c = lane * 8 + j;
        orow[c] = (v[j] - mu) * rstd * g[c] + b[c];
    }
}

// ---------------------------------------------------------------------------
// Orchestration
// ---------------------------------------------------------------------------
static inline int cdiv(long a, long b) { return (int)((a + b - 1) / b); }

extern "C" void kernel_launch(void* const* d_in, const int* in_sizes, int n_in,
                              void* d_out, int out_size, void* d_ws, size_t ws_size,
                              hipStream_t stream)
{
    const int NP = 20000, NA = 20000, E = 400000;
    const int DH = 256, T = 60000;      // tokens = NP*3

    const float* feat_p  = (const float*)d_in[0];
    const float* feat_a  = (const float*)d_in[1];
    const float* W_in_p  = (const float*)d_in[2];
    const float* b_in_p  = (const float*)d_in[3];
    const float* W_in_a  = (const float*)d_in[4];
    const float* b_in_a  = (const float*)d_in[5];
    const float* W_gap   = (const float*)d_in[6];
    const float* al_ap   = (const float*)d_in[7];
    const float* ar_ap   = (const float*)d_in[8];
    const float* b_gap   = (const float*)d_in[9];
    const float* W_gpp   = (const float*)d_in[10];
    const float* al_pp   = (const float*)d_in[11];
    const float* ar_pp   = (const float*)d_in[12];
    const float* b_gpp   = (const float*)d_in[13];
    const float* Wqkv_s  = (const float*)d_in[14];
    const float* bqkv_s  = (const float*)d_in[15];
    const float* Wo_s    = (const float*)d_in[16];
    const float* bo_s    = (const float*)d_in[17];
    const float* Wqkv_c  = (const float*)d_in[18];
    const float* bqkv_c  = (const float*)d_in[19];
    const float* Wo_c    = (const float*)d_in[20];
    const float* bo_c    = (const float*)d_in[21];
    const float* W_ff1   = (const float*)d_in[22];
    const float* b_ff1   = (const float*)d_in[23];
    const float* W_ff2   = (const float*)d_in[24];
    const float* b_ff2   = (const float*)d_in[25];
    const float* ln1_g   = (const float*)d_in[26];
    const float* ln1_b   = (const float*)d_in[27];
    const float* ln2_g   = (const float*)d_in[28];
    const float* ln2_b   = (const float*)d_in[29];
    const float* ln3_g   = (const float*)d_in[30];
    const float* ln3_b   = (const float*)d_in[31];
    const float* W_pred  = (const float*)d_in[32];
    const float* b_pred  = (const float*)d_in[33];
    const int*   ap_src  = (const int*)d_in[34];
    const int*   ap_dst  = (const int*)d_in[35];
    const int*   pp_src  = (const int*)d_in[36];
    const int*   pp_dst  = (const int*)d_in[37];

    // --- workspace layout (bytes); phase-1 and phase-2 share the arena ---
    char* ws = (char*)d_ws;
    float* TGT = (float*)ws;                                  // 60000*256 f32
    char* arena = ws + (size_t)T * DH * 4;                    // 61,440,000 B

    // phase 1 (graph) views
    float* HPA = (float*)(arena + 0);                         // 40000*256 f32 [ha; hp]
    float* Hh  = (float*)(arena + 40960000);                  // 40000*256 f32
    float* AGG = (float*)(arena + 81920000);                  // 40000*256 f32
    float* EL  = (float*)(arena + 122880000);                 // 40000*8
    float* ER  = EL + 40000 * 8;
    float* MM  = ER + 40000 * 8;
    float* ZZ  = MM + 40000 * 8;
    float* HP  = HPA + (size_t)NA * DH;                       // papers half

    // phase 2 (transformer) views (same arena, temporally disjoint)
    unsigned short* QKV = (unsigned short*)(arena + 0);       // 60000*768 bf16
    float* ATTN = (float*)(arena + 92160000);                 // 60000*256 f32
    float* MHAO = (float*)(arena + 153600000);                // 60000*256 f32
    float* XBUF = (float*)(arena + 215040000);                // 60000*256 f32
    unsigned short* FFS = (unsigned short*)(arena + 276480000); // 7680*2048 bf16
    float* FFO = (float*)(arena + 307937280);                 // 7680*256 f32

    // ---------------- phase 1: projections + GAT layers ----------------
    launch_gemm<false,true,false,false>(feat_a, W_in_a, b_in_a, HPA,
                                        NA, DH, 128, 128, 128, DH, stream);
    launch_gemm<false,true,false,false>(feat_p, W_in_p, b_in_p, HP,
                                        NP, DH, 128, 128, 128, DH, stream);
    copy_slot0<<<cdiv((long)NP * 256, 256), 256, 0, stream>>>(HP, TGT, NP);

    // GAT author->paper (homogeneous rows: [authors 0..NA) ; papers NA..)
    launch_gemm<false,false,false,false>(HPA, W_gap, nullptr, Hh,
                                         NA + NP, DH, DH, DH, DH, DH, stream);
    gat_eler<<<cdiv((long)(NA+NP)*8,256),256,0,stream>>>(Hh, al_ap, ar_ap, EL, ER, NA+NP);
    gat_init_max<<<cdiv((long)(NA+NP)*8,256),256,0,stream>>>(EL, ER, MM, NA+NP);
    gat_edge_max<<<cdiv((long)E*8,256),256,0,stream>>>(ap_src, ap_dst, NA, EL, ER, MM, E);
    gat_init_sum<<<cdiv((long)(NA+NP)*8,256),256,0,stream>>>(EL, ER, MM, Hh, ZZ, AGG, NA+NP);
    gat_edge_acc<<<cdiv((long)E*8,256),256,0,stream>>>(ap_src, ap_dst, NA, EL, ER, MM, Hh, ZZ, AGG, E);
    gat_final<<<cdiv((long)NP*8,256),256,0,stream>>>(AGG, ZZ, b_gap, TGT, NA, 1, NP);

    // GAT paper->paper
    launch_gemm<false,false,false,false>(HP, W_gpp, nullptr, Hh,
                                         NP, DH, DH, DH, DH, DH, stream);
    gat_eler<<<cdiv((long)NP*8,256),256,0,stream>>>(Hh, al_pp, ar_pp, EL, ER, NP);
    gat_init_max<<<cdiv((long)NP*8,256),256,0,stream>>>(EL, ER, MM, NP);
    gat_edge_max<<<cdiv((long)E*8,256),256,0,stream>>>(pp_src, pp_dst, 0, EL, ER, MM, E);
    gat_init_sum<<<cdiv((long)NP*8,256),256,0,stream>>>(EL, ER, MM, Hh, ZZ, AGG, NP);
    gat_edge_acc<<<cdiv((long)E*8,256),256,0,stream>>>(pp_src, pp_dst, 0, EL, ER, MM, Hh, ZZ, AGG, E);
    gat_final<<<cdiv((long)NP*8,256),256,0,stream>>>(AGG, ZZ, b_gpp, TGT, 0, 2, NP);

    // ---------------- phase 2: transformer decoder layer ----------------
    // self-attention
    launch_gemm<false,true,false,true>(TGT, Wqkv_s, bqkv_s, QKV,
                                       T, 768, DH, DH, DH, 768, stream);
    attn_s3<<<cdiv((long)NP*24,256),256,0,stream>>>(QKV, 768, QKV+256, 768, QKV+512, 768,
                                                    ATTN, NP);
    launch_gemm<false,true,false,false>(ATTN, Wo_s, bo_s, MHAO,
                                        T, DH, DH, DH, DH, DH, stream);
    ln_add<<<cdiv(T,8),256,0,stream>>>(TGT, MHAO, ln1_g, ln1_b, XBUF, T);

    // cross-attention (q from x, k/v from tgt)
    unsigned short* KVC = QKV + (size_t)T * DH;               // 60000*512 bf16
    launch_gemm<false,true,false,true>(XBUF, Wqkv_c, bqkv_c, QKV,
                                       T, DH, DH, DH, DH, DH, stream);
    launch_gemm<false,true,false,true>(TGT, Wqkv_c + 256*256, bqkv_c + 256, KVC,
                                       T, 512, DH, DH, DH, 512, stream);
    attn_s3<<<cdiv((long)NP*24,256),256,0,stream>>>(QKV, 256, KVC, 512, KVC+256, 512,
                                                    ATTN, NP);
    launch_gemm<false,true,false,false>(ATTN, Wo_c, bo_c, MHAO,
                                        T, DH, DH, DH, DH, DH, stream);
    ln_add<<<cdiv(T,8),256,0,stream>>>(XBUF, MHAO, ln2_g, ln2_b, ATTN, T);  // x2 -> ATTN

    // feed-forward, chunked over tokens (2048-wide hidden in bf16 scratch)
    const int CH = 7680;
    for (int r0 = 0; r0 < T; r0 += CH) {
        int mr = (T - r0 < CH) ? (T - r0) : CH;
        launch_gemm<false,true,true,true>(ATTN + (size_t)r0 * DH, W_ff1, b_ff1, FFS,
                                          mr, 2048, DH, DH, DH, 2048, stream);
        launch_gemm<true,true,false,false>(FFS, W_ff2, b_ff2, FFO,
                                           mr, DH, 2048, 2048, 2048, DH, stream);
        ln_add<<<cdiv(mr,8),256,0,stream>>>(ATTN + (size_t)r0 * DH, FFO, ln3_g, ln3_b,
                                            XBUF + (size_t)r0 * DH, mr);
    }

    // prediction head: z.reshape(NP, 768) @ W_pred^T + b_pred
    launch_gemm<false,true,false,false>(XBUF, W_pred, b_pred, (float*)d_out,
                                        NP, 64, 768, 768, 768, 64, stream);
}
